// MoELayer_4105988735159
// MI455X (gfx1250) — compile-verified, matrix operand order checked by reading, not run on hardware
//
#include <hip/hip_runtime.h>
#include <hip/hip_bf16.h>
#include <math.h>

// ---------------- problem constants (reference setup is fixed) --------------
constexpr int Bc  = 4;
constexpr int Tc  = 2048;
constexpr int Dc  = 1024;     // model dim
constexpr int Ec  = 8;        // experts
constexpr int FFc = 4096;     // ffn dim
constexpr int Sc  = Bc * Tc;              // 8192 tokens
constexpr int CAP = (Sc / Ec) * 5 / 4;    // 1280 (capacity factor 1.25)

typedef __bf16 bf16_t;
typedef __attribute__((ext_vector_type(8)))  __bf16 v8bf;
typedef __attribute__((ext_vector_type(16))) __bf16 v16bf;
typedef __attribute__((ext_vector_type(8)))  float  v8f;

// ---------------------------------------------------------------------------
// zero fill
__global__ void moe_zero_u32(unsigned* __restrict__ p, size_t n) {
  size_t i = (size_t)blockIdx.x * blockDim.x + threadIdx.x;
  size_t stride = (size_t)gridDim.x * blockDim.x;
  for (; i < n; i += stride) p[i] = 0u;
}

// ---------------------------------------------------------------------------
// fp32 [E][Kdim][Ndim] -> bf16 [E][Ndim][Kdim]  (transpose so K is contiguous)
__global__ void moe_convert_transpose(const float* __restrict__ W,
                                      bf16_t* __restrict__ Wt,
                                      int Kdim, int Ndim) {
  size_t total = (size_t)Ec * Kdim * Ndim;
  size_t i = (size_t)blockIdx.x * blockDim.x + threadIdx.x;
  size_t stride = (size_t)gridDim.x * blockDim.x;
  size_t plane = (size_t)Kdim * Ndim;
  for (; i < total; i += stride) {
    size_t e = i / plane;
    size_t r = i - e * plane;
    size_t k = r / (size_t)Ndim;
    size_t n = r - k * (size_t)Ndim;
    Wt[(e * (size_t)Ndim + n) * (size_t)Kdim + k] = (bf16_t)W[i];
  }
}

// ---------------------------------------------------------------------------
// gating: one wave per token. logits[e] = H[s] . gate_w[e]
// lossAcc[0] += lse^2 ; lossAcc[1+e] += softmax prob e
__global__ void moe_gate_kernel(const float* __restrict__ H,
                                const float* __restrict__ gw,
                                int* __restrict__ eidx,
                                float* __restrict__ lossAcc) {
  const int wid  = blockIdx.x * (blockDim.x >> 5) + (threadIdx.x >> 5);
  if (wid >= Sc) return;
  const int lane = threadIdx.x & 31;

  float acc[Ec];
#pragma unroll
  for (int e = 0; e < Ec; ++e) acc[e] = 0.f;

  const float* h = H + (size_t)wid * Dc;
  for (int k = lane; k < Dc; k += 32) {
    float hv = h[k];
#pragma unroll
    for (int e = 0; e < Ec; ++e) acc[e] += hv * gw[e * Dc + k];
  }
#pragma unroll
  for (int e = 0; e < Ec; ++e)
    for (int off = 16; off > 0; off >>= 1)
      acc[e] += __shfl_xor(acc[e], off, 32);

  if (lane == 0) {
    float mx = acc[0]; int arg = 0;
#pragma unroll
    for (int e = 1; e < Ec; ++e)
      if (acc[e] > mx) { mx = acc[e]; arg = e; }
    float se = 0.f;
#pragma unroll
    for (int e = 0; e < Ec; ++e) se += __expf(acc[e] - mx);
    float lse = mx + __logf(se);
    eidx[wid] = arg;
    atomicAdd(lossAcc, lse * lse);
#pragma unroll
    for (int e = 0; e < Ec; ++e) atomicAdd(lossAcc + 1 + e, __expf(acc[e] - lse));
  }
}

// ---------------------------------------------------------------------------
// token-order scan (exact cumsum semantics): single workgroup of 256 threads,
// each owning 32 consecutive tokens. Also finalizes moe_loss.
__global__ void moe_scan_kernel(const int* __restrict__ eidx,
                                int* __restrict__ pos,
                                const float* __restrict__ lossAcc,
                                float* __restrict__ lossOut) {
  __shared__ int totals[256][Ec];
  const int t = threadIdx.x;
  const int per = Sc / 256;            // 32
  const int s0 = t * per;

  int cnt[Ec];
#pragma unroll
  for (int e = 0; e < Ec; ++e) cnt[e] = 0;
  for (int i = 0; i < per; ++i) cnt[eidx[s0 + i]]++;
#pragma unroll
  for (int e = 0; e < Ec; ++e) totals[t][e] = cnt[e];
  __syncthreads();

  if (t == 0) {
    int run[Ec];
#pragma unroll
    for (int e = 0; e < Ec; ++e) run[e] = 0;
    for (int tt = 0; tt < 256; ++tt)
#pragma unroll
      for (int e = 0; e < Ec; ++e) {
        int v = totals[tt][e];
        totals[tt][e] = run[e];        // exclusive base
        run[e] += v;
      }
    // losses: router = 1e-3 * mean(lse^2); balance = 1e-2 * E * mean(load*imp)
    float router = 0.001f * lossAcc[0] / (float)Sc;
    float bal = 0.f;
#pragma unroll
    for (int e = 0; e < Ec; ++e)
      bal += ((float)run[e] / (float)Sc) * (lossAcc[1 + e] / (float)Sc);
    lossOut[0] = router + 0.01f * bal;  // E * mean_E == sum_E
  }
  __syncthreads();

  int base[Ec];
#pragma unroll
  for (int e = 0; e < Ec; ++e) base[e] = totals[t][e];
  for (int i = 0; i < per; ++i) {
    int e = eidx[s0 + i];
    int p = base[e]++;
    pos[s0 + i] = (p < CAP) ? p : -1;
  }
}

// ---------------------------------------------------------------------------
// scatter in-capacity tokens into bf16 expert buffers [E][CAP][Dc]
__global__ void moe_dispatch_kernel(const float* __restrict__ H,
                                    const int* __restrict__ eidx,
                                    const int* __restrict__ pos,
                                    bf16_t* __restrict__ buf) {
  const int s = blockIdx.x;
  const int p = pos[s];
  if (p < 0) return;
  const int e = eidx[s];
  const float* src = H + (size_t)s * Dc;
  bf16_t* dst = buf + ((size_t)e * CAP + p) * Dc;
  for (int k = threadIdx.x; k < Dc; k += blockDim.x) dst[k] = (bf16_t)src[k];
}

// ---------------------------------------------------------------------------
// WMMA fragment helpers (16-bit A/B layout per CDNA5 ISA):
// lane = (row-or-col & 15) + 16*half; element e: K = (e<8?0:16) + half*8 + (e&7)
__device__ __forceinline__ v16bf moe_load_frag(const bf16_t* __restrict__ rowK,
                                               int k0, int half) {
  const v8bf lo = *reinterpret_cast<const v8bf*>(rowK + k0 + half * 8);
  const v8bf hi = *reinterpret_cast<const v8bf*>(rowK + k0 + 16 + half * 8);
  v16bf f;
#pragma unroll
  for (int i = 0; i < 8; ++i) { f[i] = lo[i]; f[8 + i] = hi[i]; }
  return f;
}

__device__ __forceinline__ float moe_gelu(float x) {
  return 0.5f * x * (1.0f + erff(x * 0.7071067811865476f));
}

// wave-0 streams one 16x32 bf16 A chunk (1 KB) into LDS with two
// GLOBAL_LOAD_ASYNC_TO_LDS_B128 ops (32 lanes x 16 B each), ASYNCcnt-tracked.
__device__ __forceinline__ void moe_async_copy_Atile(const bf16_t* __restrict__ Aexp,
                                                     int m0, int K, int kt,
                                                     unsigned ldsDst, int lane) {
  const int row = lane >> 2;                       // 0..7
  const int seg = lane & 3;                        // 0..3 (16B segments of a row)
  const unsigned lo = ldsDst + (unsigned)(row * 64 + seg * 16);
  const uint64_t ga = (uint64_t)(uintptr_t)(Aexp + ((size_t)(m0 + row) * K + (kt << 5)))
                    + (uint64_t)(seg * 16);
  asm volatile("global_load_async_to_lds_b128 %0, %1, off"
               :: "v"(lo), "v"(ga) : "memory");
  const unsigned lo2 = lo + 512u;                  // rows 8..15
  const uint64_t ga2 = ga + (uint64_t)8 * (uint64_t)K * 2u;
  asm volatile("global_load_async_to_lds_b128 %0, %1, off"
               :: "v"(lo2), "v"(ga2) : "memory");
}

__device__ __forceinline__ void moe_wg_barrier() {
  // raw split barrier: avoids compiler-inserted full loadcnt waits, keeping
  // the register-double-buffered B prefetch in flight across iterations
  asm volatile("s_barrier_signal -1\n\ts_barrier_wait -1" ::: "memory");
}

// ---------------------------------------------------------------------------
// batched-by-expert GEMM: D[e] = A[e] (MxK row-major) * Bt[e]^T (stored [N][K])
// block = 8 waves covering one 16-row M-tile x 512 cols; A tile shared via
// double-buffered LDS (async copy by wave 0), B double-buffered in registers.
// mode 0: out = bf16 gelu(D + bias)   (GEMM1 -> h)
// mode 1: out = f32  D + bias         (GEMM2 -> out_buf)
__global__ void moe_gemm_bf16(const bf16_t* __restrict__ A,
                              const bf16_t* __restrict__ Bt,
                              const float* __restrict__ bias,
                              bf16_t* __restrict__ outBF,
                              float* __restrict__ outF,
                              int M, int N, int K, int mode) {
  __shared__ __align__(16) bf16_t Atile[2][16 * 32];   // 2 KB double buffer

  const int Mt = M >> 4;
  const int NB = N >> 9;                 // 512-wide block column tiles
  const int bid = blockIdx.x;
  const int e   = bid / (Mt * NB);
  const int r   = bid - e * (Mt * NB);
  const int m0  = (r / NB) << 4;
  const int nb0 = (r % NB) << 9;

  const int w    = threadIdx.x >> 5;
  const int lane = threadIdx.x & 31;
  const int half = lane >> 4;
  const int rc   = lane & 15;
  const int n0   = nb0 + (w << 6);       // this wave's 64-col slice
  const bool w0  = (w == 0);

  const bf16_t* Aexp = A + (size_t)e * M * K;
  const bf16_t* B0   = Bt + ((size_t)e * N + n0 + rc) * K;

  const unsigned ldsBase = (unsigned)(uintptr_t)&Atile[0][0];
  const int nk = K >> 5;

  v8f acc[4];
#pragma unroll
  for (int j = 0; j < 4; ++j)
#pragma unroll
    for (int i = 0; i < 8; ++i) acc[j][i] = 0.f;

  v16bf Bf[2][4];

  // ---- prologue: stage A tile 0, preload B fragments for k0 = 0
  if (w0) moe_async_copy_Atile(Aexp, m0, K, 0, ldsBase, lane);
#pragma unroll
  for (int j = 0; j < 4; ++j)
    Bf[0][j] = moe_load_frag(B0 + (size_t)(j * 16) * K, 0, half);

  // ---- pipelined K loop (unroll 2 so ping-pong indices constant-fold)
#pragma unroll 2
  for (int kk = 0; kk < nk; ++kk) {
    const int cur = kk & 1;
    const int nxt = cur ^ 1;

    if (w0) asm volatile("s_wait_asynccnt 0x0" ::: "memory");  // tile kk landed
    moe_wg_barrier();

    if (kk + 1 < nk) {
      if (w0)
        moe_async_copy_Atile(Aexp, m0, K, kk + 1,
                             ldsBase + (unsigned)(nxt * 1024), lane);
#pragma unroll
      for (int j = 0; j < 4; ++j)
        Bf[nxt][j] = moe_load_frag(B0 + (size_t)(j * 16) * K, (kk + 1) << 5, half);
      __builtin_prefetch(B0 + ((size_t)(kk + 2) << 5), 0, 3);
    }

    // A fragment from LDS (ds_load_b128 x2)
    const bf16_t* Ar = &Atile[cur][rc * 32];
    const v8bf alo = *reinterpret_cast<const v8bf*>(Ar + half * 8);
    const v8bf ahi = *reinterpret_cast<const v8bf*>(Ar + 16 + half * 8);
    v16bf a;
#pragma unroll
    for (int i = 0; i < 8; ++i) { a[i] = alo[i]; a[8 + i] = ahi[i]; }

#pragma unroll
    for (int j = 0; j < 4; ++j)
      acc[j] = __builtin_amdgcn_wmma_f32_16x16x32_bf16(
          /*neg_a=*/false, a, /*neg_b=*/false, Bf[cur][j],
          /*c_mod=*/(short)0, acc[j], /*reuse_a=*/false, /*reuse_b=*/false);
  }

  // ---- epilogue: lane holds D[m0 + v + 8*half][n0 + j*16 + rc], v = 0..7
  const float* be = bias + (size_t)e * N;
#pragma unroll
  for (int j = 0; j < 4; ++j) {
    const int col = n0 + j * 16 + rc;
    const float bv = be[col];
#pragma unroll
    for (int v = 0; v < 8; ++v) {
      const int row = m0 + 8 * half + v;
      const float x = acc[j][v] + bv;
      const size_t off = ((size_t)e * M + row) * (size_t)N + col;
      if (mode == 0) outBF[off] = (bf16_t)moe_gelu(x);
      else           outF[off]  = x;
    }
  }
}

// ---------------------------------------------------------------------------
// gather: out token row = out_buf[e,pos] or zeros if dropped
__global__ void moe_combine_kernel(const float* __restrict__ outbuf,
                                   const int* __restrict__ eidx,
                                   const int* __restrict__ pos,
                                   float* __restrict__ out) {
  const int s = blockIdx.x;
  const int p = pos[s];
  float* dst = out + (size_t)s * Dc;
  if (p < 0) {
    for (int k = threadIdx.x; k < Dc; k += blockDim.x) dst[k] = 0.f;
    return;
  }
  const float* src = outbuf + ((size_t)eidx[s] * CAP + p) * Dc;
  for (int k = threadIdx.x; k < Dc; k += blockDim.x) dst[k] = src[k];
}

// ---------------------------------------------------------------------------
extern "C" void kernel_launch(void* const* d_in, const int* in_sizes, int n_in,
                              void* d_out, int out_size, void* d_ws, size_t ws_size,
                              hipStream_t stream) {
  const float* H    = (const float*)d_in[0];   // [B,T,d]
  const float* gw   = (const float*)d_in[1];   // [E,d]
  const float* W1   = (const float*)d_in[2];   // [E,d,ff]
  const float* b1   = (const float*)d_in[3];   // [E,ff]
  const float* W2   = (const float*)d_in[4];   // [E,ff,d]
  const float* b2   = (const float*)d_in[5];   // [E,d]
  float* out = (float*)d_out;                  // [S*d] tokens ++ [1] loss

  // ---- workspace carve (all region sizes multiple of 256B) ----
  char* ws = (char*)d_ws;
  size_t off = 0;
  float* lossAcc = (float*)(ws + off); off += 256;                         // 1+E accumulators
  int*   eidx    = (int*)(ws + off);   off += (size_t)Sc * 4;              // 32 KB
  int*   pos     = (int*)(ws + off);   off += (size_t)Sc * 4;              // 32 KB
  bf16_t* bufB   = (bf16_t*)(ws + off); off += (size_t)Ec * CAP * Dc * 2;  // 20 MB
  bf16_t* w1t    = (bf16_t*)(ws + off); off += (size_t)Ec * FFc * Dc * 2;  // 64 MB
  bf16_t* w2t    = (bf16_t*)(ws + off); off += (size_t)Ec * Dc * FFc * 2;  // 64 MB
  bf16_t* hB     = (bf16_t*)(ws + off); off += (size_t)Ec * CAP * FFc * 2; // 80 MB
  float*  outbuf = (float*)(ws + off);  off += (size_t)Ec * CAP * Dc * 4;  // 40 MB

  // 1) zero loss accumulators and the dispatch buffer
  moe_zero_u32<<<1, 64, 0, stream>>>((unsigned*)lossAcc, 16);
  {
    size_t nwords = (size_t)Ec * CAP * Dc * 2 / 4;
    moe_zero_u32<<<2048, 256, 0, stream>>>((unsigned*)bufB, nwords);
  }

  // 2) weight convert + transpose to K-contiguous bf16
  moe_convert_transpose<<<4096, 256, 0, stream>>>(W1, w1t, Dc, FFc);
  moe_convert_transpose<<<4096, 256, 0, stream>>>(W2, w2t, FFc, Dc);

  // 3) gating (one wave32 per token; 8 waves per 256-thread block)
  moe_gate_kernel<<<Sc / 8, 256, 0, stream>>>(H, gw, eidx, lossAcc);

  // 4) token-order scan + loss finalize into d_out[S*d]
  moe_scan_kernel<<<1, 256, 0, stream>>>(eidx, pos, lossAcc, out + (size_t)Sc * Dc);

  // 5) scatter tokens into expert buffers
  moe_dispatch_kernel<<<Sc, 256, 0, stream>>>(H, eidx, pos, bufB);

  // 6) GEMM1: [E,CAP,Dc] x [E,Dc,FFc] + b1, GELU -> bf16 h
  {
    int blocks = Ec * (CAP / 16) * (FFc / 512);   // 5120 blocks x 8 waves
    moe_gemm_bf16<<<blocks, 256, 0, stream>>>(bufB, w1t, b1, hB, nullptr,
                                              CAP, FFc, Dc, 0);
  }
  // 7) GEMM2: [E,CAP,FFc] x [E,FFc,Dc] + b2 -> f32 out_buf
  {
    int blocks = Ec * (CAP / 16) * (Dc / 512);    // 1280 blocks x 8 waves
    moe_gemm_bf16<<<blocks, 256, 0, stream>>>(hB, w2t, b2, nullptr, outbuf,
                                              CAP, Dc, FFc, 1);
  }

  // 8) gather back (zeros for dropped tokens; no gate weighting, per reference)
  moe_combine_kernel<<<Sc, 256, 0, stream>>>(outbuf, eidx, pos, out);
}
